// SMoE_14061722927308
// MI455X (gfx1250) — matrix operand order repaired and checked by reading, not compiled, not run on hardware
//
#include <hip/hip_runtime.h>
#include <hip/hip_bf16.h>
#include <math.h>

// ---------------- problem constants ----------------
#define DIM   1024
#define HID   4096
#define NEXP  8
#define NTOK  8192          // 4 * 2048
#define CAP   1280          // int(NTOK/NEXP * 1.25)
#define ALPHAF 0.01f

typedef __bf16 bf16_t;
typedef __attribute__((ext_vector_type(16))) __bf16 v16bf;
typedef __attribute__((ext_vector_type(8)))  float  v8f;

// =====================================================================
// WMMA 16x16x32 bf16 fragment loaders (layouts per CDNA5 ISA 7.12.2)
// =====================================================================
// A: 16(M) x 32(K). lane<16: M=lane, K {0..7} in V0-3, {16..23} in V4-7.
//                   lane>=16: M=lane-16, K {8..15} and {24..31}.
__device__ __forceinline__ v16bf load_a_bf16(const bf16_t* __restrict__ base, int ld) {
  const int lane = threadIdx.x & 31;
  const int m    = lane & 15;
  const int k0   = (lane >> 4) << 3;          // 0 or 8
  const bf16_t* p = base + (size_t)m * ld + k0;
  union { v16bf v; uint4 q[2]; } u;
  u.q[0] = *(const uint4*)(p);                // K = k0   .. k0+7
  u.q[1] = *(const uint4*)(p + 16);           // K = k0+16.. k0+23
  return u.v;
}

// B: 32(K) x 16(N). lanes 0-15 hold column N=lane, K {0..7} and {16..23};
//                   lanes 16-31 hold same columns, K {8..15} and {24..31}.
// Source: weight row n, K-contiguous, fp32 -> convert to bf16 in regs.
__device__ __forceinline__ v16bf load_b_f32(const float* __restrict__ base, int ld) {
  const int lane = threadIdx.x & 31;
  const int n    = lane & 15;
  const int k0   = (lane >> 4) << 3;          // 0 or 8
  const float* p = base + (size_t)n * ld + k0;
  float4 f0 = *(const float4*)(p);
  float4 f1 = *(const float4*)(p + 4);
  float4 f2 = *(const float4*)(p + 16);
  float4 f3 = *(const float4*)(p + 20);
  v16bf r;
  r[0]=(__bf16)f0.x; r[1]=(__bf16)f0.y; r[2]=(__bf16)f0.z; r[3]=(__bf16)f0.w;
  r[4]=(__bf16)f1.x; r[5]=(__bf16)f1.y; r[6]=(__bf16)f1.z; r[7]=(__bf16)f1.w;
  r[8]=(__bf16)f2.x; r[9]=(__bf16)f2.y; r[10]=(__bf16)f2.z; r[11]=(__bf16)f2.w;
  r[12]=(__bf16)f3.x; r[13]=(__bf16)f3.y; r[14]=(__bf16)f3.z; r[15]=(__bf16)f3.w;
  return r;
}

#define WMMA_BF16(A,B,C) __builtin_amdgcn_wmma_f32_16x16x32_bf16( \
    false, (A), false, (B), (short)0, (C), false, false)

// =====================================================================
// 0) zero the tiny atomic accumulators
// =====================================================================
__global__ void moe_init_kernel(int* __restrict__ cnt, float* __restrict__ psum) {
  int i = threadIdx.x;
  if (i < NEXP) { cnt[i] = 0; psum[i] = 0.0f; }
}

// =====================================================================
// 1) gating: logits, argmax expert, softmax gate weight, f_i/p_i stats
//    one block (256 thr) per token
// =====================================================================
__global__ __launch_bounds__(256) void moe_gate_kernel(
    const float* __restrict__ x, const float* __restrict__ Wg,
    const float* __restrict__ bg, int* __restrict__ eidx,
    float* __restrict__ gw, int* __restrict__ cnt, float* __restrict__ psum) {
  const int t   = blockIdx.x;
  const int tid = threadIdx.x;
  const float4 xv = *(const float4*)(x + (size_t)t * DIM + tid * 4);
  float p[NEXP];
#pragma unroll
  for (int e = 0; e < NEXP; ++e) {
    const float4 wv = *(const float4*)(Wg + (size_t)e * DIM + tid * 4);
    p[e] = xv.x*wv.x + xv.y*wv.y + xv.z*wv.z + xv.w*wv.w;
  }
#pragma unroll
  for (int e = 0; e < NEXP; ++e)
#pragma unroll
    for (int s = 16; s > 0; s >>= 1) p[e] += __shfl_down(p[e], s, 32);

  __shared__ float sm[8][NEXP];
  __shared__ float logits[NEXP];
  const int wave = tid >> 5, lane = tid & 31;
  if (lane == 0)
#pragma unroll
    for (int e = 0; e < NEXP; ++e) sm[wave][e] = p[e];
  __syncthreads();
  if (tid < NEXP) {
    float s = bg[tid];
#pragma unroll
    for (int w = 0; w < 8; ++w) s += sm[w][tid];
    logits[tid] = s;
  }
  __syncthreads();
  if (tid == 0) {
    int best = 0; float bestv = logits[0];
#pragma unroll
    for (int e = 1; e < NEXP; ++e)
      if (logits[e] > bestv) { bestv = logits[e]; best = e; }
    float den = 0.0f, we[NEXP];
#pragma unroll
    for (int e = 0; e < NEXP; ++e) { we[e] = __expf(logits[e] - bestv); den += we[e]; }
    const float inv = 1.0f / den;
    eidx[t] = best;
    gw[t]   = we[best] * inv;
    atomicAdd(&cnt[best], 1);
#pragma unroll
    for (int e = 0; e < NEXP; ++e) atomicAdd(&psum[e], we[e] * inv);
  }
}

// =====================================================================
// 2) ordered per-expert positions (matches reference cumsum semantics)
//    one block per expert, 1024-thread Hillis-Steele scan over chunks
// =====================================================================
__global__ __launch_bounds__(1024) void moe_pos_kernel(
    const int* __restrict__ eidx, int* __restrict__ pos) {
  const int e   = blockIdx.x;
  const int tid = threadIdx.x;
  __shared__ int sd[1024];
  __shared__ int sbase;
  if (tid == 0) sbase = 0;
  __syncthreads();
  for (int c = 0; c < NTOK; c += 1024) {
    const int t    = c + tid;
    const int flag = (eidx[t] == e) ? 1 : 0;
    sd[tid] = flag;
    __syncthreads();
    for (int off = 1; off < 1024; off <<= 1) {
      int v = (tid >= off) ? sd[tid - off] : 0;
      __syncthreads();
      sd[tid] += v;
      __syncthreads();
    }
    if (flag) {
      const int pp = sbase + sd[tid] - 1;
      pos[t] = (pp < CAP) ? pp : -1;
    }
    __syncthreads();
    if (tid == 0) sbase += sd[1023];
    __syncthreads();
  }
}

// =====================================================================
// 3) scatter tokens fp32 -> bf16 expert buffers
// =====================================================================
__global__ __launch_bounds__(256) void moe_scatter_kernel(
    const float* __restrict__ x, const int* __restrict__ eidx,
    const int* __restrict__ pos, bf16_t* __restrict__ buf) {
  const int t = blockIdx.x;
  const int p = pos[t];
  if (p < 0) return;
  const int e   = eidx[t];
  const int tid = threadIdx.x;
  const float4 xv = *(const float4*)(x + (size_t)t * DIM + tid * 4);
  bf16_t* dst = buf + ((size_t)e * CAP + p) * DIM + tid * 4;
  union { ushort4 us; bf16_t b[4]; } u;
  u.b[0] = (__bf16)xv.x; u.b[1] = (__bf16)xv.y;
  u.b[2] = (__bf16)xv.z; u.b[3] = (__bf16)xv.w;
  *(ushort4*)dst = u.us;
}

// =====================================================================
// 4) fused GEMM1: H = silu(X @ W2^T) * (X @ W3^T)   (per expert)
//    M=CAP, N=HID, K=DIM. WG 256thr = 8 waves -> 64x128 tile,
//    wave tile 32x32, dual accumulators share A fragments.
// =====================================================================
__global__ __launch_bounds__(256) void moe_gemm1_kernel(
    const bf16_t* __restrict__ buf, const float* __restrict__ W2,
    const float* __restrict__ W3, bf16_t* __restrict__ H) {
  const int e  = blockIdx.z;
  const int m0 = blockIdx.y * 64 + ((threadIdx.x >> 5) & 1) * 32;
  const int n0 = blockIdx.x * 128 + ((threadIdx.x >> 5) >> 1) * 32;
  const bf16_t* A  = buf + (size_t)e * CAP * DIM + (size_t)m0 * DIM;
  const float*  B2 = W2  + (size_t)e * HID * DIM + (size_t)n0 * DIM;
  const float*  B3 = W3  + (size_t)e * HID * DIM + (size_t)n0 * DIM;

  v8f z = {0.f,0.f,0.f,0.f,0.f,0.f,0.f,0.f};
  v8f acc2[2][2] = {{z,z},{z,z}};
  v8f acc3[2][2] = {{z,z},{z,z}};

  for (int k = 0; k < DIM; k += 32) {
    __builtin_prefetch(B2 + k + 256, 0, 1);
    __builtin_prefetch(B3 + k + 256, 0, 1);
    v16bf a0  = load_a_bf16(A + k, DIM);
    v16bf a1  = load_a_bf16(A + (size_t)16 * DIM + k, DIM);
    v16bf b20 = load_b_f32(B2 + k, DIM);
    v16bf b21 = load_b_f32(B2 + (size_t)16 * DIM + k, DIM);
    v16bf b30 = load_b_f32(B3 + k, DIM);
    v16bf b31 = load_b_f32(B3 + (size_t)16 * DIM + k, DIM);
    acc2[0][0] = WMMA_BF16(a0, b20, acc2[0][0]);
    acc2[0][1] = WMMA_BF16(a0, b21, acc2[0][1]);
    acc2[1][0] = WMMA_BF16(a1, b20, acc2[1][0]);
    acc2[1][1] = WMMA_BF16(a1, b21, acc2[1][1]);
    acc3[0][0] = WMMA_BF16(a0, b30, acc3[0][0]);
    acc3[0][1] = WMMA_BF16(a0, b31, acc3[0][1]);
    acc3[1][0] = WMMA_BF16(a1, b30, acc3[1][0]);
    acc3[1][1] = WMMA_BF16(a1, b31, acc3[1][1]);
  }

  const int lane = threadIdx.x & 31;
  const int half = lane >> 4;
  const int nc   = lane & 15;
  bf16_t* Hp = H + (size_t)e * CAP * HID;
#pragma unroll
  for (int mi = 0; mi < 2; ++mi)
#pragma unroll
    for (int ni = 0; ni < 2; ++ni)
#pragma unroll
      for (int v = 0; v < 8; ++v) {
        const float s2 = acc2[mi][ni][v];
        const float s3 = acc3[mi][ni][v];
        const float hv = (s2 / (1.0f + __expf(-s2))) * s3;   // silu(s2)*s3
        const int row = m0 + mi * 16 + v + half * 8;
        const int col = n0 + ni * 16 + nc;
        Hp[(size_t)row * HID + col] = (__bf16)hv;
      }
}

// =====================================================================
// 5) GEMM2: EO = H @ W1^T  (per expert). M=CAP, N=DIM, K=HID.
// =====================================================================
__global__ __launch_bounds__(256) void moe_gemm2_kernel(
    const bf16_t* __restrict__ H, const float* __restrict__ W1,
    float* __restrict__ EO) {
  const int e  = blockIdx.z;
  const int m0 = blockIdx.y * 64 + ((threadIdx.x >> 5) & 1) * 32;
  const int n0 = blockIdx.x * 128 + ((threadIdx.x >> 5) >> 1) * 32;
  const bf16_t* A = H  + (size_t)e * CAP * HID + (size_t)m0 * HID;
  const float*  B = W1 + (size_t)e * DIM * HID + (size_t)n0 * HID;

  v8f z = {0.f,0.f,0.f,0.f,0.f,0.f,0.f,0.f};
  v8f acc[2][2] = {{z,z},{z,z}};

  for (int k = 0; k < HID; k += 32) {
    __builtin_prefetch(B + k + 256, 0, 1);
    v16bf a0 = load_a_bf16(A + k, HID);
    v16bf a1 = load_a_bf16(A + (size_t)16 * HID + k, HID);
    v16bf b0 = load_b_f32(B + k, HID);
    v16bf b1 = load_b_f32(B + (size_t)16 * HID + k, HID);
    acc[0][0] = WMMA_BF16(a0, b0, acc[0][0]);
    acc[0][1] = WMMA_BF16(a0, b1, acc[0][1]);
    acc[1][0] = WMMA_BF16(a1, b0, acc[1][0]);
    acc[1][1] = WMMA_BF16(a1, b1, acc[1][1]);
  }

  const int lane = threadIdx.x & 31;
  const int half = lane >> 4;
  const int nc   = lane & 15;
  float* Ep = EO + (size_t)e * CAP * DIM;
#pragma unroll
  for (int mi = 0; mi < 2; ++mi)
#pragma unroll
    for (int ni = 0; ni < 2; ++ni)
#pragma unroll
      for (int v = 0; v < 8; ++v) {
        const int row = m0 + mi * 16 + v + half * 8;
        const int col = n0 + ni * 16 + nc;
        Ep[(size_t)row * DIM + col] = acc[mi][ni][v];
      }
}

// =====================================================================
// 6) gather * gate weight (dropped tokens -> 0)
// =====================================================================
__global__ __launch_bounds__(256) void moe_gather_kernel(
    const float* __restrict__ EO, const int* __restrict__ eidx,
    const int* __restrict__ pos, const float* __restrict__ gw,
    float* __restrict__ out) {
  const int t   = blockIdx.x;
  const int tid = threadIdx.x;
  float4 r = {0.f, 0.f, 0.f, 0.f};
  const int p = pos[t];
  if (p >= 0) {
    const int e  = eidx[t];
    const float g = gw[t];
    const float4 v = *(const float4*)(EO + ((size_t)e * CAP + p) * DIM + tid * 4);
    r.x = v.x * g; r.y = v.y * g; r.z = v.z * g; r.w = v.w * g;
  }
  *(float4*)(out + (size_t)t * DIM + tid * 4) = r;
}

// =====================================================================
// 7) aux loss scalar
// =====================================================================
__global__ void moe_aux_kernel(const int* __restrict__ cnt,
                               const float* __restrict__ psum,
                               float* __restrict__ out_aux) {
  if (threadIdx.x == 0 && blockIdx.x == 0) {
    float s = 0.0f;
#pragma unroll
    for (int e = 0; e < NEXP; ++e) s += (float)cnt[e] * psum[e];
    out_aux[0] = ALPHAF * (float)NEXP * s / ((float)NTOK * (float)NTOK);
  }
}

// =====================================================================
// launcher
// =====================================================================
extern "C" void kernel_launch(void* const* d_in, const int* in_sizes, int n_in,
                              void* d_out, int out_size, void* d_ws, size_t ws_size,
                              hipStream_t stream) {
  const float* x  = (const float*)d_in[0];
  const float* Wg = (const float*)d_in[1];
  const float* bg = (const float*)d_in[2];
  const float* W1 = (const float*)d_in[3];
  const float* W2 = (const float*)d_in[4];
  const float* W3 = (const float*)d_in[5];
  float* out = (float*)d_out;

  char* ws = (char*)d_ws;
  size_t off = 0;
  auto take = [&](size_t bytes) -> char* {
    char* p = ws + off;
    off = (off + bytes + 255) & ~(size_t)255;
    return p;
  };
  int*    eidx = (int*)   take((size_t)NTOK * 4);
  int*    pos  = (int*)   take((size_t)NTOK * 4);
  float*  gwp  = (float*) take((size_t)NTOK * 4);
  int*    cnt  = (int*)   take(NEXP * 4);
  float*  psum = (float*) take(NEXP * 4);
  bf16_t* buf  = (bf16_t*)take((size_t)NEXP * CAP * DIM * 2);
  bf16_t* Hbuf = (bf16_t*)take((size_t)NEXP * CAP * HID * 2);
  float*  EO   = (float*) take((size_t)NEXP * CAP * DIM * 4);

  moe_init_kernel<<<1, 64, 0, stream>>>(cnt, psum);
  moe_gate_kernel<<<NTOK, 256, 0, stream>>>(x, Wg, bg, eidx, gwp, cnt, psum);
  moe_pos_kernel<<<NEXP, 1024, 0, stream>>>(eidx, pos);
  moe_scatter_kernel<<<NTOK, 256, 0, stream>>>(x, eidx, pos, buf);
  moe_gemm1_kernel<<<dim3(HID / 128, CAP / 64, NEXP), 256, 0, stream>>>(buf, W2, W3, Hbuf);
  moe_gemm2_kernel<<<dim3(DIM / 128, CAP / 64, NEXP), 256, 0, stream>>>(Hbuf, W1, EO);
  moe_gather_kernel<<<NTOK, 256, 0, stream>>>(EO, eidx, pos, gwp, out);
  moe_aux_kernel<<<1, 32, 0, stream>>>(cnt, psum, out + (size_t)NTOK * DIM);
}
